// AtomicConvLayer_26405458935845
// MI455X (gfx1250) — matrix operand order repaired and checked by reading, not compiled, not run on hardware
//
#include <hip/hip_runtime.h>

// AtomicConvLayer for MI455X (gfx1250), wave32, WMMA bf16 path.
//
// Restructured math (exact):
//   P     = atom @ msg_W1[:D]  + msg_b1                      (G1, per atom)
//   hs[n] = sum_m relu(P[n] + atom[idx[n,m]] @ msg_W1[D:])   (edge kernel)
//   agg   = hs @ msg_W2 + 32*msg_b2                          (G2)
//   u     = relu(atom @ upd_W1[:D] + agg @ upd_W1[D:] + upd_b1)  (G3, K=256)
//   out   = relu(atom + u @ upd_W2 + upd_b2)                 (G4)
//
// All GEMMs use V_WMMA_F32_16X16X32_BF16 (fp32 accumulate).
// Edge gather staged memory->LDS with GLOBAL_LOAD_ASYNC_TO_LDS_B128 (ASYNCcnt).
// Workspace use: ~32.3 MB.

typedef __attribute__((ext_vector_type(16))) __bf16 v16bf;
typedef __attribute__((ext_vector_type(8)))  float  v8f;

union ABPack { uint4 q[2]; v16bf v; };   // 32 bytes = 16 bf16 per lane

__device__ __forceinline__ unsigned short f2bf(float f) {
  unsigned int u = __float_as_uint(f);
  unsigned int r = u + 0x7FFFu + ((u >> 16) & 1u);   // round-to-nearest-even
  return (unsigned short)(r >> 16);
}

// ---------------------------------------------------------------- convert
__global__ void cvt_f32_bf16_kernel(const float* __restrict__ in,
                                    unsigned short* __restrict__ out, int n) {
  int i = blockIdx.x * blockDim.x + threadIdx.x;
  if (i < n) out[i] = f2bf(in[i]);
}

// ---------------------------------------------------------------- weight pack
// Packs a 128x128 fp32 slice (rows kBase..kBase+127 of a row-major [K,128] W)
// into the per-lane WMMA B-operand layout: per (kt,nt) tile, lane l holds 16
// consecutive bf16 = B[k][n] with n = nt*16 + (l&15), k = kt*32 + (l>>4)*8 + (j&7) + (j>>3)*16.
__global__ void pack_w_kernel(const float* __restrict__ W, int kBase,
                              unsigned short* __restrict__ out) {
  int p = blockIdx.x * blockDim.x + threadIdx.x;   // 0 .. 16383
  int j    = p & 15;
  int lane = (p >> 4) & 31;
  int tile = p >> 9;            // kt*8 + nt, 0..31
  int nt = tile & 7, kt = tile >> 3;
  int k = kt * 32 + (lane >> 4) * 8 + (j & 7) + (j >> 3) * 16;
  int n = nt * 16 + (lane & 15);
  out[p] = f2bf(W[(size_t)(kBase + k) * 128 + n]);
}

// ---------------------------------------------------------------- generic GEMM
// C[16 rows x 128 cols] = A0@B0 (+ A1@B1) + bias*biasScale (+residual) [relu]
// Block: 256 thr = 8 waves; wave w owns N-tile w. A is row-major bf16 [rows,128].
__global__ void gemm16_kernel(const unsigned short* __restrict__ A0,
                              const unsigned short* __restrict__ B0P,
                              const unsigned short* __restrict__ A1,
                              const unsigned short* __restrict__ B1P,
                              const float* __restrict__ bias, float biasScale,
                              const float* __restrict__ residual, int doRelu,
                              float* __restrict__ outF32,
                              unsigned short* __restrict__ outBF16,
                              int numRows) {
  int r0 = blockIdx.x * 16;
  int t = threadIdx.x, w = t >> 5, l = t & 31;
  int half = l >> 4, ln = l & 15;
  int arow = r0 + ln; if (arow >= numRows) arow = numRows - 1;   // clamp loads

  v8f acc = {};
  const uint4* a0p = (const uint4*)(A0 + (size_t)arow * 128);
  const uint4* b0p = (const uint4*)B0P;
#pragma unroll
  for (int kt = 0; kt < 4; ++kt) {
    int c = kt * 32 + half * 8;                 // A lane K-chunks: c..c+7, c+16..c+23
    ABPack a, b;
    a.q[0] = a0p[c >> 3];
    a.q[1] = a0p[(c >> 3) + 2];
    size_t bo = ((size_t)(kt * 8 + w) * 32 + l) * 2;
    b.q[0] = b0p[bo]; b.q[1] = b0p[bo + 1];
    acc = __builtin_amdgcn_wmma_f32_16x16x32_bf16(false, a.v, false, b.v,
                                                  (short)0, acc, false, false);
  }
  if (A1) {                                     // second K=128 half (concat input)
    const uint4* a1p = (const uint4*)(A1 + (size_t)arow * 128);
    const uint4* b1p = (const uint4*)B1P;
#pragma unroll
    for (int kt = 0; kt < 4; ++kt) {
      int c = kt * 32 + half * 8;
      ABPack a, b;
      a.q[0] = a1p[c >> 3];
      a.q[1] = a1p[(c >> 3) + 2];
      size_t bo = ((size_t)(kt * 8 + w) * 32 + l) * 2;
      b.q[0] = b1p[bo]; b.q[1] = b1p[bo + 1];
      acc = __builtin_amdgcn_wmma_f32_16x16x32_bf16(false, a.v, false, b.v,
                                                    (short)0, acc, false, false);
    }
  }
  // Epilogue. C layout: lane l -> col = l&15 (of this N-tile), VGPR v -> row v + 8*(l>>4).
  int col = w * 16 + ln;
  float bs = bias[col] * biasScale;
#pragma unroll
  for (int v = 0; v < 8; ++v) {
    int grow = r0 + v + 8 * half;
    if (grow < numRows) {
      float val = acc[v] + bs;
      if (residual) val += residual[(size_t)grow * 128 + col];
      if (doRelu)   val = fmaxf(val, 0.f);
      if (outF32)  outF32[(size_t)grow * 128 + col] = val;
      if (outBF16) outBF16[(size_t)grow * 128 + col] = f2bf(val);
    }
  }
}

// ---------------------------------------------------------------- edge kernel
// Two atoms per block. The 2x32 gathered neighbor rows (16 KB bf16) are staged
// memory->LDS with async-to-LDS b128 copies (ASYNCcnt), then each of 8 waves
// owns (atom = w>>2, N-tile pair = w&3): A regs reused across 2 B tiles ->
// 4 WMMAs per K-step, 16 per wave. Fused relu(P + .)-rowsum epilogue -> hs.
__global__ void edge_kernel(const unsigned short* __restrict__ atomBf,
                            const int* __restrict__ nbrIdx,           // [N,32]
                            const unsigned short* __restrict__ W1botP,
                            const float* __restrict__ P,              // [N,128]
                            unsigned short* __restrict__ hsBf,
                            int numAtoms) {
  __shared__ __align__(16) unsigned short ldsA[2 * 32 * 128];   // 16 KB
  __shared__ int ldsIdx[64];
  int n0 = blockIdx.x * 2, t = threadIdx.x;

  if (t < 64) {
    int aa = t >> 5, m = t & 31;
    int n = n0 + aa; if (n >= numAtoms) n = numAtoms - 1;
    ldsIdx[t] = nbrIdx[(size_t)n * 32 + m];
  }
  __syncthreads();

  // Async gather: 1024 x 16B chunks, 256 threads x 4. Chunk c -> LDS offset
  // c*16 B = atom (c>>9), row (c>>4)&31, col-chunk (c&15)*8 bf16.
  unsigned ldsBase = (unsigned)(size_t)(void*)ldsA;
#pragma unroll
  for (int it = 0; it < 4; ++it) {
    int c = it * 256 + t;
    int aa = c >> 9, row = (c >> 4) & 31, ch = c & 15;
    unsigned long long g = (unsigned long long)(size_t)
        (atomBf + (size_t)ldsIdx[aa * 32 + row] * 128 + ch * 8);
    unsigned laddr = ldsBase + (unsigned)c * 16;
    asm volatile("global_load_async_to_lds_b128 %0, %1, off"
                 :: "v"(laddr), "v"(g) : "memory");
  }
  asm volatile("s_wait_asynccnt 0x0" ::: "memory");
  __syncthreads();

  int w = t >> 5, l = t & 31;
  int half = l >> 4, ln = l & 15;
  int aa = w >> 2, ntp = w & 3;            // atom slot, N-tile pair
  int n = n0 + aa;
  const unsigned short* As = ldsA + aa * 32 * 128;
  const uint4* bp = (const uint4*)W1botP;

  v8f acc[2][2] = {};                       // [M-tile][N-tile]
#pragma unroll
  for (int kt = 0; kt < 4; ++kt) {
    int c = kt * 32 + half * 8;
    ABPack a0, a1, b0, b1;
    a0.q[0] = *(const uint4*)(As + ln * 128 + c);
    a0.q[1] = *(const uint4*)(As + ln * 128 + c + 16);
    a1.q[0] = *(const uint4*)(As + (16 + ln) * 128 + c);
    a1.q[1] = *(const uint4*)(As + (16 + ln) * 128 + c + 16);
    size_t bo0 = ((size_t)(kt * 8 + ntp * 2) * 32 + l) * 2;
    size_t bo1 = ((size_t)(kt * 8 + ntp * 2 + 1) * 32 + l) * 2;
    b0.q[0] = bp[bo0]; b0.q[1] = bp[bo0 + 1];
    b1.q[0] = bp[bo1]; b1.q[1] = bp[bo1 + 1];
    acc[0][0] = __builtin_amdgcn_wmma_f32_16x16x32_bf16(false, a0.v, false, b0.v,
                                                        (short)0, acc[0][0], false, false);
    acc[0][1] = __builtin_amdgcn_wmma_f32_16x16x32_bf16(false, a0.v, false, b1.v,
                                                        (short)0, acc[0][1], false, false);
    acc[1][0] = __builtin_amdgcn_wmma_f32_16x16x32_bf16(false, a1.v, false, b0.v,
                                                        (short)0, acc[1][0], false, false);
    acc[1][1] = __builtin_amdgcn_wmma_f32_16x16x32_bf16(false, a1.v, false, b1.v,
                                                        (short)0, acc[1][1], false, false);
  }

  // relu(P + C), reduce over 32 neighbor rows (16 here + 16 in lane l^16).
  if (n < numAtoms) {                        // uniform per wave
#pragma unroll
    for (int j = 0; j < 2; ++j) {
      int col = ntp * 32 + j * 16 + ln;
      float pp = P[(size_t)n * 128 + col];
      float partial = 0.f;
#pragma unroll
      for (int v = 0; v < 8; ++v) {
        partial += fmaxf(acc[0][j][v] + pp, 0.f);
        partial += fmaxf(acc[1][j][v] + pp, 0.f);
      }
      partial += __shfl_xor(partial, 16, 32);
      if (half == 0) hsBf[(size_t)n * 128 + col] = f2bf(partial);
    }
  }
}

// ---------------------------------------------------------------- launch
extern "C" void kernel_launch(void* const* d_in, const int* in_sizes, int n_in,
                              void* d_out, int out_size, void* d_ws, size_t ws_size,
                              hipStream_t stream) {
  const float* atomF  = (const float*)d_in[0];
  // d_in[1] = nbr_features: unused by the reference forward.
  const int*   nbrIdx = (const int*)d_in[2];     // harness delivers integer inputs as int32
  const float* msgW1  = (const float*)d_in[3];   // [256,128]
  const float* msgB1  = (const float*)d_in[4];
  const float* msgW2  = (const float*)d_in[5];   // [128,128]
  const float* msgB2  = (const float*)d_in[6];
  const float* updW1  = (const float*)d_in[7];   // [256,128]
  const float* updB1  = (const float*)d_in[8];
  const float* updW2  = (const float*)d_in[9];   // [128,128]
  const float* updB2  = (const float*)d_in[10];
  float* out = (float*)d_out;

  const int N = in_sizes[0] / 128;               // 25000

  // ---- workspace carve (256B aligned) ----
  uintptr_t p = (uintptr_t)d_ws;
  auto carve = [&](size_t bytes) -> void* {
    uintptr_t q = (p + 255) & ~(uintptr_t)255;
    p = q + bytes;
    return (void*)q;
  };
  unsigned short* atomBf = (unsigned short*)carve((size_t)N * 128 * 2);
  unsigned short* hsBf   = (unsigned short*)carve((size_t)N * 128 * 2);
  unsigned short* aggBf  = (unsigned short*)carve((size_t)N * 128 * 2);
  float*          Pbuf   = (float*)         carve((size_t)N * 128 * 4);
  unsigned short* uBf    = (unsigned short*)Pbuf;   // alias: P dead before u is written
  unsigned short* packs  = (unsigned short*)carve((size_t)6 * 128 * 128 * 2);
  unsigned short* pW1top = packs;
  unsigned short* pW1bot = packs + 16384;
  unsigned short* pW2    = packs + 2 * 16384;
  unsigned short* pU1top = packs + 3 * 16384;
  unsigned short* pU1bot = packs + 4 * 16384;
  unsigned short* pU2    = packs + 5 * 16384;

  // ---- stage 0: conversions + weight packing ----
  int nelA = N * 128;
  cvt_f32_bf16_kernel<<<(nelA + 255) / 256, 256, 0, stream>>>(atomF, atomBf, nelA);
  pack_w_kernel<<<64, 256, 0, stream>>>(msgW1, 0,   pW1top);
  pack_w_kernel<<<64, 256, 0, stream>>>(msgW1, 128, pW1bot);
  pack_w_kernel<<<64, 256, 0, stream>>>(msgW2, 0,   pW2);
  pack_w_kernel<<<64, 256, 0, stream>>>(updW1, 0,   pU1top);
  pack_w_kernel<<<64, 256, 0, stream>>>(updW1, 128, pU1bot);
  pack_w_kernel<<<64, 256, 0, stream>>>(updW2, 0,   pU2);

  int gemmBlocks = (N + 15) / 16;

  // ---- G1: P = atom @ W1_top + b1 ----
  gemm16_kernel<<<gemmBlocks, 256, 0, stream>>>(
      atomBf, pW1top, nullptr, nullptr, msgB1, 1.0f,
      nullptr, 0, Pbuf, nullptr, N);

  // ---- edge: hs = sum_m relu(P + gather @ W1_bot), 2 atoms/block ----
  edge_kernel<<<(N + 1) / 2, 256, 0, stream>>>(atomBf, nbrIdx, pW1bot, Pbuf, hsBf, N);

  // ---- G2: agg = hs @ W2 + 32*b2 ----
  gemm16_kernel<<<gemmBlocks, 256, 0, stream>>>(
      hsBf, pW2, nullptr, nullptr, msgB2, 32.0f,
      nullptr, 0, nullptr, aggBf, N);

  // ---- G3: u = relu(atom @ U1_top + agg @ U1_bot + ub1) ----
  gemm16_kernel<<<gemmBlocks, 256, 0, stream>>>(
      atomBf, pU1top, aggBf, pU1bot, updB1, 1.0f,
      nullptr, 1, nullptr, uBf, N);

  // ---- G4: out = relu(atom + u @ U2 + ub2) ----
  gemm16_kernel<<<gemmBlocks, 256, 0, stream>>>(
      uBf, pU2, nullptr, nullptr, updB2, 1.0f,
      atomF, 1, out, nullptr, N);
}
